// DifferentialAttention_14920716386995
// MI455X (gfx1250) — compile-verified, matrix operand order checked by reading, not run on hardware
//
#include <hip/hip_runtime.h>
#include <math.h>
#include <stdint.h>

// ---------------------------------------------------------------------------
// DiffTransformer block for MI455X (gfx1250): bf16 WMMA everywhere,
// async global->LDS staging (ASYNCcnt) with double buffering in the GEMMs.
// B=2, S=2048, E=1024, H=16, split-head d=32, value head dv=128.
// ---------------------------------------------------------------------------

typedef __attribute__((ext_vector_type(16))) __bf16 v16bf;
typedef __attribute__((ext_vector_type(8)))  __bf16 v8bf;
typedef __attribute__((ext_vector_type(8)))  float  v8f;
typedef unsigned short u16;
typedef unsigned int   u32;

#define S_LEN 2048
#define BATCH 2
#define NHEAD 16
#define DH    32      // per-head dim of Q1/Q2/K1/K2
#define DV    128     // per-head value dim (2*D)
#define LDT   40      // LDS tile leading dim (ushorts); multiple of 8 -> 16B rows
#define TILE_ELTS (128 * LDT)          // one GEMM stage buffer, in ushorts
#define TILE_BYTES (TILE_ELTS * 2)

__device__ __forceinline__ u16 f2bf(float f) {
    union { float f; u32 u; } c; c.f = f;
    u32 r = c.u + 0x7FFFu + ((c.u >> 16) & 1u);   // round-to-nearest-even
    return (u16)(r >> 16);
}
__device__ __forceinline__ float bf2f(u16 b) {
    union { u32 u; float f; } c; c.u = ((u32)b) << 16;
    return c.f;
}

// A- and B-fragment loader for v_wmma_f32_16x16x32_bf16.
// Memory layout requirement: 16 rows (M for A, N for B), K contiguous, ld elems.
// Per ISA 7.12.2: lane L -> row = L%16; lanes 0-15 hold K=[0..7]+[16..23],
// lanes 16-31 hold K=[8..15]+[24..31].
template <typename P>
__device__ __forceinline__ v16bf load_frag(P base, int ld) {
    const int lane = threadIdx.x & 31;
    const int row  = lane & 15;
    const int kof  = (lane >> 4) * 8;
    const v8bf* p0 = (const v8bf*)(&base[row * ld + kof]);
    const v8bf* p1 = (const v8bf*)(&base[row * ld + kof + 16]);
    v8bf lo = *p0, hi = *p1;
    v16bf r;
#pragma unroll
    for (int i = 0; i < 8; i++) { r[i] = lo[i]; r[8 + i] = hi[i]; }
    return r;
}

__device__ __forceinline__ v8f wmma_bf16(v16bf a, v16bf b, v8f c) {
    return __builtin_amdgcn_wmma_f32_16x16x32_bf16(false, a, false, b,
                                                   (short)0, c, false, false);
}

// Async copy of 32 contiguous bytes global -> LDS ("GV" mode, ASYNCcnt).
__device__ __forceinline__ void async_cp32(u32 ldsAddr, const u16* g) {
    unsigned long long ga = (unsigned long long)(uintptr_t)g;
    asm volatile("global_load_async_to_lds_b128 %0, %1, off\n\t"
                 "global_load_async_to_lds_b128 %0, %1, off offset:16"
                 :: "v"(ldsAddr), "v"(ga) : "memory");
}

// ---------------------------------------------------------------------------
// Shared 128x128 GEMM mainloop: C[m,n] = sum_k A[m,k]*B[n,k]  (both K-major).
// 256 threads = 8 waves in a 2x4 grid; each wave owns 64x32, i.e. 4x2 tiles.
// Double-buffered async global->LDS staging: per K-step each thread issues
// 4 async b128 ops for the NEXT tile, then waits s_wait_asynccnt 4 so only
// the current tile's (in-order) completions gate compute.
// ---------------------------------------------------------------------------
__device__ __forceinline__ void gemm128(const u16* __restrict__ Ag,
                                        const u16* __restrict__ Bg,
                                        int K, int m0, int n0,
                                        u16* sA0, u16* sB0, v8f acc[4][2]) {
    const int tid   = threadIdx.x;
    const int wid   = tid >> 5;
    const int wm    = wid >> 2;    // 0..1
    const int wn    = wid & 3;     // 0..3
    const int crow  = tid >> 1;    // 0..127
    const int chalf = tid & 1;     // 16 elems each

    const u16* aSrc = Ag + (size_t)(m0 + crow) * K + chalf * 16;
    const u16* bSrc = Bg + (size_t)(n0 + crow) * K + chalf * 16;
    const u32 aDst0 = (u32)(uintptr_t)(sA0 + crow * LDT + chalf * 16);
    const u32 bDst0 = (u32)(uintptr_t)(sB0 + crow * LDT + chalf * 16);
    const u32 aDst1 = aDst0 + TILE_BYTES;
    const u32 bDst1 = bDst0 + TILE_BYTES;

    const int NT = K >> 5;
    // prologue: stage tile 0 into buffer 0
    async_cp32(aDst0, aSrc);
    async_cp32(bDst0, bSrc);

    for (int t = 0; t < NT; ++t) {
        const int buf = t & 1;
        if (t + 1 < NT) {
            async_cp32(buf ? aDst0 : aDst1, aSrc + (t + 1) * 32);
            async_cp32(buf ? bDst0 : bDst1, bSrc + (t + 1) * 32);
            asm volatile("s_wait_asynccnt 0x4" ::: "memory");
        } else {
            asm volatile("s_wait_asynccnt 0x0" ::: "memory");
        }
        __syncthreads();

        const u16* cA = sA0 + buf * TILE_ELTS;
        const u16* cB = sB0 + buf * TILE_ELTS;
        v16bf af[4], bfr[2];
#pragma unroll
        for (int mi = 0; mi < 4; mi++)
            af[mi] = load_frag(cA + (wm * 64 + mi * 16) * LDT, LDT);
#pragma unroll
        for (int ni = 0; ni < 2; ni++)
            bfr[ni] = load_frag(cB + (wn * 32 + ni * 16) * LDT, LDT);
#pragma unroll
        for (int mi = 0; mi < 4; mi++)
#pragma unroll
            for (int ni = 0; ni < 2; ni++)
                acc[mi][ni] = wmma_bf16(af[mi], bfr[ni], acc[mi][ni]);
        __syncthreads();
    }
}

// ---------------------------------------------------------------------------
// Elementwise fp32 -> bf16 cast
// ---------------------------------------------------------------------------
__global__ void __launch_bounds__(256) k_cast(const float* __restrict__ src,
                                              u16* __restrict__ dst, int n) {
    int i = blockIdx.x * 256 + threadIdx.x;
    if (i < n) dst[i] = f2bf(src[i]);
}

// ---------------------------------------------------------------------------
// lambda = exp(dot(lq1,lk1)) - exp(dot(lq2,lk2)) + LAMBDA_INIT(=0.2)
// ---------------------------------------------------------------------------
__global__ void k_lambda(const float* q1, const float* q2,
                         const float* k1, const float* k2, float* lam) {
    __shared__ float s1[64], s2[64];
    int t = threadIdx.x;
    s1[t] = q1[t] * k1[t];
    s2[t] = q2[t] * k2[t];
    __syncthreads();
    for (int st = 32; st > 0; st >>= 1) {
        if (t < st) { s1[t] += s1[t + st]; s2[t] += s2[t + st]; }
        __syncthreads();
    }
    if (t == 0) lam[0] = __expf(s1[0]) - __expf(s2[0]) + 0.2f;
}

// ---------------------------------------------------------------------------
// Fused QKV GEMM: M=4096 (B*S), N=4096 (Q|K|V channels), K=1024.
// n<2048 -> QK buffer row-major; n>=2048 -> V stored transposed [bh][dv][s].
// ---------------------------------------------------------------------------
__global__ void __launch_bounds__(256) k_gemm_qkv(const u16* __restrict__ Xb,
                                                  const u16* __restrict__ Wb,
                                                  u16* __restrict__ QK,
                                                  u16* __restrict__ Vt) {
    __shared__ u16 sA[2 * TILE_ELTS];
    __shared__ u16 sB[2 * TILE_ELTS];
    v8f z = {0, 0, 0, 0, 0, 0, 0, 0};
    v8f acc[4][2];
#pragma unroll
    for (int i = 0; i < 4; i++)
#pragma unroll
        for (int j = 0; j < 2; j++) acc[i][j] = z;

    const int m0 = blockIdx.y * 128, n0 = blockIdx.x * 128;
    gemm128(Xb, Wb, 1024, m0, n0, sA, sB, acc);

    const int lane = threadIdx.x & 31, wid = threadIdx.x >> 5;
    const int wm = wid >> 2, wn = wid & 3;
    const int colL = lane & 15, rbase = (lane >> 4) * 8;
#pragma unroll
    for (int mi = 0; mi < 4; mi++)
#pragma unroll
        for (int ni = 0; ni < 2; ni++)
#pragma unroll
            for (int r = 0; r < 8; r++) {
                int m = m0 + wm * 64 + mi * 16 + rbase + r;
                int n = n0 + wn * 32 + ni * 16 + colL;
                u16 v = f2bf(acc[mi][ni][r]);
                if (n < 2048) {
                    QK[(size_t)m * 2048 + n] = v;
                } else {
                    int c = n - 2048, h = c >> 7, dv = c & 127;
                    int b = m >> 11, s = m & 2047;
                    Vt[((size_t)(b * NHEAD + h) * DV + dv) * S_LEN + s] = v;
                }
            }
}

// ---------------------------------------------------------------------------
// RoPE (d=16 rotation pairs per 32-wide half-head) + Q scale (D^-0.5 = 1/8).
// Scatters into [bh][s][32] layouts for Q1/Q2/K1/K2.
// ---------------------------------------------------------------------------
__global__ void __launch_bounds__(256) k_rope(const u16* __restrict__ QK,
                                              u16* __restrict__ Q1h, u16* __restrict__ Q2h,
                                              u16* __restrict__ K1h, u16* __restrict__ K2h) {
    u32 idx = blockIdx.x * 256 + threadIdx.x;   // B*S*H*16*4 = 4194304
    int d = idx & 15;   idx >>= 4;
    int h = idx & 15;   idx >>= 4;
    int s = idx & 2047; idx >>= 11;
    int b = idx & 1;    idx >>= 1;
    int t = idx;        // 0:Q1 1:Q2 2:K1 3:K2
    int m = b * 2048 + s;
    int base = (t & 1) * 512 + (t >> 1) * 1024 + h * 32;
    float x1 = bf2f(QK[(size_t)m * 2048 + base + d]);
    float x2 = bf2f(QK[(size_t)m * 2048 + base + d + 16]);
    float inv = __powf(10000.f, -(float)d * (1.f / 256.f));
    float ang = (float)s * inv;
    float c = __cosf(ang), sn = __sinf(ang);
    float y1 = x1 * c - x2 * sn;
    float y2 = x2 * c + x1 * sn;
    float sc = (t < 2) ? 0.125f : 1.f;
    u16* dst = (t == 0) ? Q1h : (t == 1) ? Q2h : (t == 2) ? K1h : K2h;
    size_t o = ((size_t)(b * NHEAD + h) * S_LEN + s) * DH + d;
    dst[o]      = f2bf(y1 * sc);
    dst[o + 16] = f2bf(y2 * sc);
}

// ---------------------------------------------------------------------------
// Differential flash attention. Block = 4 waves, each wave owns 16 queries.
// Per 32-key step: 4 score WMMAs + 16 P.V WMMAs; online softmax via shfl_xor.
// ---------------------------------------------------------------------------
__global__ void __launch_bounds__(128) k_attn(const u16* __restrict__ Q1h,
                                              const u16* __restrict__ Q2h,
                                              const u16* __restrict__ K1h,
                                              const u16* __restrict__ K2h,
                                              const u16* __restrict__ Vt,
                                              const float* __restrict__ lamp,
                                              float* __restrict__ O) {
    __shared__ u16 sP[4][16 * LDT];
    const int lane = threadIdx.x & 31;
    const int wid  = threadIdx.x >> 5;
    const int bh   = blockIdx.y;
    const int q0   = blockIdx.x * 64 + wid * 16;
    const float lam = lamp[0];
    const int rbase = (lane >> 4) * 8, colL = lane & 15;

    v16bf a1 = load_frag(Q1h + ((size_t)bh * S_LEN + q0) * DH, DH);
    v16bf a2 = load_frag(Q2h + ((size_t)bh * S_LEN + q0) * DH, DH);

    v8f z = {0, 0, 0, 0, 0, 0, 0, 0};
    v8f o1[8], o2[8];
    float m1[8], l1[8], m2[8], l2[8];
#pragma unroll
    for (int j = 0; j < 8; j++) { o1[j] = z; o2[j] = z; }
#pragma unroll
    for (int r = 0; r < 8; r++) { m1[r] = -1e30f; l1[r] = 0.f; m2[r] = -1e30f; l2[r] = 0.f; }

    u16* lp = sP[wid];
    const u16* vbase = Vt + (size_t)bh * DV * S_LEN;

    for (int kt = 0; kt < S_LEN; kt += 32) {
        const u16* k1p = K1h + ((size_t)bh * S_LEN + kt) * DH;
        const u16* k2p = K2h + ((size_t)bh * S_LEN + kt) * DH;
        if (kt + 32 < S_LEN) {   // near-scope prefetch of next key tile
            __builtin_prefetch(k1p + 32 * DH, 0, 3);
            __builtin_prefetch(k2p + 32 * DH, 0, 3);
        }
        v16bf b10 = load_frag(k1p, DH);
        v16bf b11 = load_frag(k1p + 16 * DH, DH);
        v16bf b20 = load_frag(k2p, DH);
        v16bf b21 = load_frag(k2p + 16 * DH, DH);
        v8f s10 = wmma_bf16(a1, b10, z);
        v8f s11 = wmma_bf16(a1, b11, z);
        v8f s20 = wmma_bf16(a2, b20, z);
        v8f s21 = wmma_bf16(a2, b21, z);

        // ---- attn1: online softmax update + stage P1 in LDS ----
        float corr[8];
#pragma unroll
        for (int r = 0; r < 8; r++) {
            float v0 = s10[r], v1 = s11[r];
            float mx = fmaxf(v0, v1);
            mx = fmaxf(mx, __shfl_xor(mx, 1));
            mx = fmaxf(mx, __shfl_xor(mx, 2));
            mx = fmaxf(mx, __shfl_xor(mx, 4));
            mx = fmaxf(mx, __shfl_xor(mx, 8));
            float mn = fmaxf(m1[r], mx);
            float cr = __expf(m1[r] - mn);
            float p0 = __expf(v0 - mn), p1 = __expf(v1 - mn);
            float rs = p0 + p1;
            rs += __shfl_xor(rs, 1); rs += __shfl_xor(rs, 2);
            rs += __shfl_xor(rs, 4); rs += __shfl_xor(rs, 8);
            l1[r] = l1[r] * cr + rs;
            m1[r] = mn;
            corr[r] = cr;
            int row = rbase + r;
            lp[row * LDT + colL]      = f2bf(p0);
            lp[row * LDT + colL + 16] = f2bf(p1);
        }
#pragma unroll
        for (int j = 0; j < 8; j++)
#pragma unroll
            for (int r = 0; r < 8; r++) o1[j][r] *= corr[r];
        asm volatile("s_wait_dscnt 0" ::: "memory");
        {
            v16bf ap = load_frag(lp, LDT);
#pragma unroll
            for (int j = 0; j < 8; j++) {
                v16bf bv = load_frag(vbase + (size_t)(j * 16) * S_LEN + kt, S_LEN);
                o1[j] = wmma_bf16(ap, bv, o1[j]);
            }
        }

        // ---- attn2 ----
#pragma unroll
        for (int r = 0; r < 8; r++) {
            float v0 = s20[r], v1 = s21[r];
            float mx = fmaxf(v0, v1);
            mx = fmaxf(mx, __shfl_xor(mx, 1));
            mx = fmaxf(mx, __shfl_xor(mx, 2));
            mx = fmaxf(mx, __shfl_xor(mx, 4));
            mx = fmaxf(mx, __shfl_xor(mx, 8));
            float mn = fmaxf(m2[r], mx);
            float cr = __expf(m2[r] - mn);
            float p0 = __expf(v0 - mn), p1 = __expf(v1 - mn);
            float rs = p0 + p1;
            rs += __shfl_xor(rs, 1); rs += __shfl_xor(rs, 2);
            rs += __shfl_xor(rs, 4); rs += __shfl_xor(rs, 8);
            l2[r] = l2[r] * cr + rs;
            m2[r] = mn;
            corr[r] = cr;
            int row = rbase + r;
            lp[row * LDT + colL]      = f2bf(p0);
            lp[row * LDT + colL + 16] = f2bf(p1);
        }
#pragma unroll
        for (int j = 0; j < 8; j++)
#pragma unroll
            for (int r = 0; r < 8; r++) o2[j][r] *= corr[r];
        asm volatile("s_wait_dscnt 0" ::: "memory");
        {
            v16bf ap = load_frag(lp, LDT);
#pragma unroll
            for (int j = 0; j < 8; j++) {
                v16bf bv = load_frag(vbase + (size_t)(j * 16) * S_LEN + kt, S_LEN);
                o2[j] = wmma_bf16(ap, bv, o2[j]);
            }
        }
    }

    // out = O1/l1 - lam * O2/l2, stored [bh][s][dv] fp32
#pragma unroll
    for (int j = 0; j < 8; j++)
#pragma unroll
        for (int r = 0; r < 8; r++) {
            int row = q0 + rbase + r;
            int dv  = j * 16 + colL;
            float val = o1[j][r] / l1[r] - lam * (o2[j][r] / l2[r]);
            O[((size_t)bh * S_LEN + row) * DV + dv] = val;
        }
}

// ---------------------------------------------------------------------------
// Group-norm stats: one block per (b,h), reduce over S*DV = 262144 elements.
// ---------------------------------------------------------------------------
__global__ void __launch_bounds__(256) k_gnstats(const float* __restrict__ O,
                                                 float* __restrict__ stats) {
    __shared__ float ssum[256], ssq[256];
    const int bh = blockIdx.x;
    const float* p = O + (size_t)bh * (S_LEN * DV);
    float s = 0.f, q = 0.f;
    for (int i = threadIdx.x; i < S_LEN * DV; i += 256) {
        float v = p[i]; s += v; q += v * v;
    }
    ssum[threadIdx.x] = s; ssq[threadIdx.x] = q;
    __syncthreads();
    for (int st = 128; st > 0; st >>= 1) {
        if (threadIdx.x < st) {
            ssum[threadIdx.x] += ssum[threadIdx.x + st];
            ssq[threadIdx.x]  += ssq[threadIdx.x + st];
        }
        __syncthreads();
    }
    if (threadIdx.x == 0) {
        float n = (float)(S_LEN * DV);
        float mu = ssum[0] / n;
        float var = ssq[0] / n - mu * mu;
        stats[bh * 2]     = mu;
        stats[bh * 2 + 1] = rsqrtf(var + 1e-5f);
    }
}

// ---------------------------------------------------------------------------
// Normalize + (1-LAMBDA_INIT)=0.8 scale + the reference's reshape/transpose
// permutation, emitting the projection-GEMM input X2[b][s'][c] in bf16.
// ---------------------------------------------------------------------------
__global__ void __launch_bounds__(256) k_gnnorm(const float* __restrict__ O,
                                                const float* __restrict__ stats,
                                                u16* __restrict__ X2) {
    size_t idx = (size_t)blockIdx.x * 256 + threadIdx.x;   // B*H*S*DV = 2^23
    int b  = (int)(idx >> 22);           // H*S*DV = 2^22
    int bh = (int)(idx >> 18);           // S*DV   = 2^18
    size_t f = idx & 4194303u;           // flat (h*S+s)*128+d within batch
    float mu = stats[bh * 2], rs = stats[bh * 2 + 1];
    float v = (O[idx] - mu) * rs * 0.8f;
    int c  = (int)(f >> 11);             // reshape (b, 2048, 2048)
    int sp = (int)(f & 2047);            // then transpose(0,2,1)
    X2[((size_t)b * 2048 + sp) * 2048 + c] = f2bf(v);
}

// ---------------------------------------------------------------------------
// Projection GEMM: M=4096, N=2048, K=2048, fp32 output.
// ---------------------------------------------------------------------------
__global__ void __launch_bounds__(256) k_gemm_proj(const u16* __restrict__ X2,
                                                   const u16* __restrict__ Wb,
                                                   float* __restrict__ P) {
    __shared__ u16 sA[2 * TILE_ELTS];
    __shared__ u16 sB[2 * TILE_ELTS];
    v8f z = {0, 0, 0, 0, 0, 0, 0, 0};
    v8f acc[4][2];
#pragma unroll
    for (int i = 0; i < 4; i++)
#pragma unroll
        for (int j = 0; j < 2; j++) acc[i][j] = z;

    const int m0 = blockIdx.y * 128, n0 = blockIdx.x * 128;
    gemm128(X2, Wb, 2048, m0, n0, sA, sB, acc);

    const int lane = threadIdx.x & 31, wid = threadIdx.x >> 5;
    const int wm = wid >> 2, wn = wid & 3;
    const int colL = lane & 15, rbase = (lane >> 4) * 8;
#pragma unroll
    for (int mi = 0; mi < 4; mi++)
#pragma unroll
        for (int ni = 0; ni < 2; ni++)
#pragma unroll
            for (int r = 0; r < 8; r++) {
                int m = m0 + wm * 64 + mi * 16 + rbase + r;
                int n = n0 + wn * 32 + ni * 16 + colL;
                P[(size_t)m * 2048 + n] = acc[mi][ni][r];
            }
}

// ---------------------------------------------------------------------------
// arctan gate: out = gate(a) * v, a = p[:, :1024], v = p[:, 1024:]
// ---------------------------------------------------------------------------
__global__ void __launch_bounds__(256) k_gate(const float* __restrict__ P,
                                              const float* __restrict__ alpha,
                                              float* __restrict__ out) {
    size_t idx = (size_t)blockIdx.x * 256 + threadIdx.x;   // B*S*1024
    int j = (int)(idx & 1023);
    size_t row = idx >> 10;
    float a = P[row * 2048 + j];
    float v = P[row * 2048 + 1024 + j];
    float al = alpha[0];
    float g = (atanf(a) + 1.57079632679f) * 0.31830988618f;
    g = g * (1.f + 2.f * al) - al;
    out[idx] = g * v;
}

// ---------------------------------------------------------------------------
extern "C" void kernel_launch(void* const* d_in, const int* in_sizes, int n_in,
                              void* d_out, int out_size, void* d_ws, size_t ws_size,
                              hipStream_t stream) {
    (void)in_sizes; (void)n_in; (void)out_size; (void)ws_size;
    const float* x   = (const float*)d_in[0];
    const float* Wq  = (const float*)d_in[1];
    const float* Wk  = (const float*)d_in[2];
    const float* Wv  = (const float*)d_in[3];
    const float* lq1 = (const float*)d_in[4];
    const float* lq2 = (const float*)d_in[5];
    const float* lk1 = (const float*)d_in[6];
    const float* lk2 = (const float*)d_in[7];
    const float* Wp  = (const float*)d_in[8];
    const float* alp = (const float*)d_in[9];
    float* out = (float*)d_out;

    char* w = (char*)d_ws;
    size_t off = 0;
    auto alloc = [&](size_t bytes) -> void* {
        void* p = w + off;
        off += (bytes + 255) & ~(size_t)255;
        return p;
    };
    u16*   Xb    = (u16*)  alloc((size_t)4096 * 1024 * 2);       // x bf16
    u16*   Wqkv  = (u16*)  alloc((size_t)4096 * 1024 * 2);       // [Wq;Wk;Wv] bf16
    u16*   Wpb   = (u16*)  alloc((size_t)2048 * 2048 * 2);       // Wproj bf16
    u16*   QK    = (u16*)  alloc((size_t)4096 * 2048 * 2);       // Q|K row-major
    u16*   Vt    = (u16*)  alloc((size_t)BATCH * NHEAD * DV * S_LEN * 2); // V^T per head
    u16*   Q1h   = (u16*)  alloc((size_t)BATCH * NHEAD * S_LEN * DH * 2);
    u16*   Q2h   = (u16*)  alloc((size_t)BATCH * NHEAD * S_LEN * DH * 2);
    u16*   K1h   = (u16*)  alloc((size_t)BATCH * NHEAD * S_LEN * DH * 2);
    u16*   K2h   = (u16*)  alloc((size_t)BATCH * NHEAD * S_LEN * DH * 2);
    float* lam   = (float*)alloc(256);
    float* Obuf  = (float*)alloc((size_t)BATCH * NHEAD * S_LEN * DV * 4);
    float* stats = (float*)alloc(256);
    u16*   X2    = (u16*)  alloc((size_t)BATCH * 2048 * 2048 * 2);
    float* Pbuf  = (float*)alloc((size_t)BATCH * 2048 * 2048 * 4);

    // 1. bf16 casts
    k_cast<<<16384, 256, 0, stream>>>(x,  Xb,            4096 * 1024);
    k_cast<<< 4096, 256, 0, stream>>>(Wq, Wqkv,          1024 * 1024);
    k_cast<<< 4096, 256, 0, stream>>>(Wk, Wqkv + 1024 * 1024, 1024 * 1024);
    k_cast<<< 8192, 256, 0, stream>>>(Wv, Wqkv + 2048 * 1024, 2048 * 1024);
    k_cast<<<16384, 256, 0, stream>>>(Wp, Wpb,           2048 * 2048);

    // 2. lambda scalar
    k_lambda<<<1, 64, 0, stream>>>(lq1, lq2, lk1, lk2, lam);

    // 3. fused QKV GEMM (M=4096, N=4096, K=1024)
    k_gemm_qkv<<<dim3(32, 32), 256, 0, stream>>>(Xb, Wqkv, QK, Vt);

    // 4. RoPE + Q scaling, head-major layouts
    k_rope<<<16384, 256, 0, stream>>>(QK, Q1h, Q2h, K1h, K2h);

    // 5. differential flash attention
    k_attn<<<dim3(S_LEN / 64, BATCH * NHEAD), 128, 0, stream>>>(
        Q1h, Q2h, K1h, K2h, Vt, lam, Obuf);

    // 6. group norm
    k_gnstats<<<BATCH * NHEAD, 256, 0, stream>>>(Obuf, stats);
    k_gnnorm<<<32768, 256, 0, stream>>>(Obuf, stats, X2);

    // 7. output projection (M=4096, N=2048, K=2048)
    k_gemm_proj<<<dim3(16, 32), 256, 0, stream>>>(X2, Wpb, Pbuf);

    // 8. arctan gate
    k_gate<<<16384, 256, 0, stream>>>(Pbuf, alp, out);
}
